// MaskedAttention_37469294691214
// MI455X (gfx1250) — compile-verified, matrix operand order checked by reading, not run on hardware
//
#include <hip/hip_runtime.h>
#include <hip/hip_bf16.h>

typedef __attribute__((ext_vector_type(16))) __bf16 v16bf;
typedef __attribute__((ext_vector_type(8)))  __bf16 bf16x8;
typedef __attribute__((ext_vector_type(8)))  float  v8f;

#define B_  2
#define T_  2048
#define C_  1024
#define H_  16
#define D_  64

__device__ __forceinline__ v8f wmma_bf16(v16bf a, v16bf b, v8f c) {
  return __builtin_amdgcn_wmma_f32_16x16x32_bf16(false, a, false, b, (short)0, c, false, false);
}

// Load one 16x32 (A) or 32x16-as-columns (B) bf16 fragment from a row-major
// [rows, stride] buffer. Per the CDNA5 16-bit fragment layout:
//   lane l -> row r0 + (l&15); K elements k0 + (l>>4)*8 + {0..7} and +16+{0..7}
__device__ __forceinline__ v16bf load_frag_g(const __bf16* __restrict__ base,
                                             int stride, int r0, int k0, int lane) {
  int r  = r0 + (lane & 15);
  int kb = k0 + ((lane >> 4) << 3);
  const __bf16* p = base + (size_t)r * (size_t)stride + kb;
  bf16x8 lo = *(const bf16x8*)(p);
  bf16x8 hi = *(const bf16x8*)(p + 16);
  return __builtin_shufflevector(lo, hi, 0,1,2,3,4,5,6,7,8,9,10,11,12,13,14,15);
}

// ---------------------------------------------------------------- convert ---
__global__ __launch_bounds__(256) void cvt_kernel(const float* __restrict__ x,
                                                  const float* __restrict__ Wqkv,
                                                  const float* __restrict__ Wo,
                                                  __bf16* __restrict__ xb,
                                                  __bf16* __restrict__ WqkvT,
                                                  __bf16* __restrict__ WoT) {
  int id = blockIdx.x * 256 + threadIdx.x;
  if (id < B_ * T_ * C_) xb[id] = (__bf16)x[id];
  if (id < 3 * C_ * C_) {                 // WqkvT[n][k] = Wqkv[k][n]
    int n = id >> 10, k = id & 1023;
    WqkvT[id] = (__bf16)Wqkv[(size_t)k * (3 * C_) + n];
  }
  if (id < C_ * C_) {                     // WoT[n][k] = Wo[k][n]
    int n = id >> 10, k = id & 1023;
    WoT[id] = (__bf16)Wo[(size_t)k * C_ + n];
  }
}

// --------------------------------------------------------------- QKV GEMM ---
// [4096,1024] x [1024,3072] + bqkv. Wave tile 64x64 (4x4 WMMA accumulators:
// 8 fragment loads feed 16 WMMAs per K=32 step). Block 8 waves = 128x256.
// Epilogue scatters Q[B,H,T,D] (pre-scaled 1/8), K[B,H,T,D], Vt[B,H,D,T], bf16.
__global__ __launch_bounds__(256) void qkv_gemm(const __bf16* __restrict__ xb,
                                                const __bf16* __restrict__ WT,
                                                const float*  __restrict__ bias,
                                                __bf16* __restrict__ Qb,
                                                __bf16* __restrict__ Kb,
                                                __bf16* __restrict__ Vt) {
  int lane = threadIdx.x & 31;
  int w    = threadIdx.x >> 5;
  int m0 = blockIdx.y * 128 + (w & 1) * 64;
  int n0 = blockIdx.x * 256 + (w >> 1) * 64;
  v8f acc[4][4] = {};
  for (int k0 = 0; k0 < C_; k0 += 32) {
    v16bf a[4], b[4];
#pragma unroll
    for (int i = 0; i < 4; ++i) a[i] = load_frag_g(xb, C_, m0 + i * 16, k0, lane);
#pragma unroll
    for (int j = 0; j < 4; ++j) b[j] = load_frag_g(WT, C_, n0 + j * 16, k0, lane);
#pragma unroll
    for (int i = 0; i < 4; ++i)
#pragma unroll
      for (int j = 0; j < 4; ++j)
        acc[i][j] = wmma_bf16(a[i], b[j], acc[i][j]);
  }
  int nl = lane & 15, half8 = (lane >> 4) << 3;
#pragma unroll
  for (int i = 0; i < 4; ++i)
#pragma unroll
    for (int j = 0; j < 4; ++j)
#pragma unroll
      for (int v = 0; v < 8; ++v) {
        int row = m0 + i * 16 + half8 + v;
        int col = n0 + j * 16 + nl;
        float val = acc[i][j][v] + bias[col];
        int b2 = row >> 11, t = row & (T_ - 1);
        if (col < C_) {
          int h = col >> 6, d = col & 63;
          Qb[(((size_t)(b2 * H_ + h)) * T_ + t) * D_ + d] = (__bf16)(val * 0.125f);
        } else if (col < 2 * C_) {
          int c = col - C_; int h = c >> 6, d = c & 63;
          Kb[(((size_t)(b2 * H_ + h)) * T_ + t) * D_ + d] = (__bf16)val;
        } else {
          int c = col - 2 * C_; int h = c >> 6, d = c & 63;
          Vt[(((size_t)(b2 * H_ + h)) * D_ + d) * T_ + t] = (__bf16)val;
        }
      }
}

// ---------------------------------------------------------- flash attention -
// One wave per 16-row Q tile of one (b,h). Streams 32-key KV tiles with
// online softmax; P re-layout C-format -> A-fragment via padded LDS tile.
__global__ __launch_bounds__(256) void attn_kernel(const __bf16* __restrict__ Qb,
                                                   const __bf16* __restrict__ Kb,
                                                   const __bf16* __restrict__ Vt,
                                                   __bf16* __restrict__ ao) {
  __shared__ __align__(16) __bf16 pbuf[8 * 16 * 40];   // 8 waves x 16x32, stride 40
  int lane = threadIdx.x & 31;
  int w    = threadIdx.x >> 5;
  int gw = blockIdx.x * 8 + w;          // 0..4095
  int bh = gw >> 7;                     // (b*H + h)
  int qt = gw & 127;
  int q0 = qt << 4;
  const __bf16* Q = Qb + (size_t)bh * T_ * D_;
  const __bf16* K = Kb + (size_t)bh * T_ * D_;
  const __bf16* V = Vt + (size_t)bh * D_ * T_;

  v16bf qf0 = load_frag_g(Q, D_, q0, 0,  lane);
  v16bf qf1 = load_frag_g(Q, D_, q0, 32, lane);

  v8f o0 = {}, o1 = {}, o2 = {}, o3 = {};
  float mrun[8], lrun[8];
#pragma unroll
  for (int v = 0; v < 8; ++v) { mrun[v] = -3.0e38f; lrun[v] = 0.0f; }
  int nl = lane & 15, half8 = (lane >> 4) << 3;
  __bf16* pb = &pbuf[w * 640];

  int qEnd = q0 + 15;
  for (int kv = 0; kv <= qEnd; kv += 32) {
    // S = Q x K^T  (two 16x16 f32 tiles: keys kv..+15, kv+16..+31)
    v8f s0 = {}, s1 = {};
    s0 = wmma_bf16(qf0, load_frag_g(K, D_, kv,      0,  lane), s0);
    s0 = wmma_bf16(qf1, load_frag_g(K, D_, kv,      32, lane), s0);
    s1 = wmma_bf16(qf0, load_frag_g(K, D_, kv + 16, 0,  lane), s1);
    s1 = wmma_bf16(qf1, load_frag_g(K, D_, kv + 16, 32, lane), s1);
    // causal mask (only on diagonal-crossing tiles)
    if (kv + 31 > q0) {
#pragma unroll
      for (int v = 0; v < 8; ++v) {
        int q = q0 + half8 + v;
        if (kv + nl      > q) s0[v] = -3.0e38f;
        if (kv + 16 + nl > q) s1[v] = -3.0e38f;
      }
    }
    // online softmax: rows align with accumulator VGPR index
#pragma unroll
    for (int v = 0; v < 8; ++v) {
      float rmax = fmaxf(s0[v], s1[v]);
#pragma unroll
      for (int m = 1; m < 16; m <<= 1) rmax = fmaxf(rmax, __shfl_xor(rmax, m));
      float nm    = fmaxf(mrun[v], rmax);
      float alpha = __expf(mrun[v] - nm);
      float p0 = __expf(s0[v] - nm);
      float p1 = __expf(s1[v] - nm);
      float rs = p0 + p1;
#pragma unroll
      for (int m = 1; m < 16; m <<= 1) rs += __shfl_xor(rs, m);
      lrun[v] = lrun[v] * alpha + rs;
      mrun[v] = nm;
      o0[v] *= alpha; o1[v] *= alpha; o2[v] *= alpha; o3[v] *= alpha;
      s0[v] = p0; s1[v] = p1;
    }
    // C-layout probs -> LDS (row m = half8+v, col = key-in-tile)
#pragma unroll
    for (int v = 0; v < 8; ++v) {
      int m = half8 + v;
      pb[m * 40 + nl]      = (__bf16)s0[v];
      pb[m * 40 + 16 + nl] = (__bf16)s1[v];
    }
    __asm volatile("" ::: "memory");
    // re-read as A-fragment (16x32): lane -> row nl, K chunk at half8 / +16
    const __bf16* pr = pb + nl * 40 + half8;
    bf16x8 plo = *(const bf16x8*)(pr);
    bf16x8 phi = *(const bf16x8*)(pr + 16);
    v16bf pf = __builtin_shufflevector(plo, phi, 0,1,2,3,4,5,6,7,8,9,10,11,12,13,14,15);
    // O += P x V   (V^T rows are d, contiguous along keys)
    o0 = wmma_bf16(pf, load_frag_g(V, T_, 0,  kv, lane), o0);
    o1 = wmma_bf16(pf, load_frag_g(V, T_, 16, kv, lane), o1);
    o2 = wmma_bf16(pf, load_frag_g(V, T_, 32, kv, lane), o2);
    o3 = wmma_bf16(pf, load_frag_g(V, T_, 48, kv, lane), o3);
    __asm volatile("" ::: "memory");
  }
  // normalize and write [B*T, C] bf16 for the output projection
  int b = bh >> 4, h = bh & (H_ - 1);
#pragma unroll
  for (int v = 0; v < 8; ++v) {
    float inv = 1.0f / lrun[v];
    int row = (b << 11) + q0 + half8 + v;
    size_t base = (size_t)row * C_ + h * D_ + nl;
    ao[base]      = (__bf16)(o0[v] * inv);
    ao[base + 16] = (__bf16)(o1[v] * inv);
    ao[base + 32] = (__bf16)(o2[v] * inv);
    ao[base + 48] = (__bf16)(o3[v] * inv);
  }
}

// ------------------------------------------------------------- output GEMM --
__global__ __launch_bounds__(256) void out_gemm(const __bf16* __restrict__ ao,
                                                const __bf16* __restrict__ WoT,
                                                const float*  __restrict__ bo,
                                                float* __restrict__ out) {
  int lane = threadIdx.x & 31;
  int w    = threadIdx.x >> 5;
  int m0 = blockIdx.y * 128 + (w & 1) * 64;
  int n0 = blockIdx.x * 256 + (w >> 1) * 64;
  v8f acc[4][4] = {};
  for (int k0 = 0; k0 < C_; k0 += 32) {
    v16bf a[4], b[4];
#pragma unroll
    for (int i = 0; i < 4; ++i) a[i] = load_frag_g(ao,  C_, m0 + i * 16, k0, lane);
#pragma unroll
    for (int j = 0; j < 4; ++j) b[j] = load_frag_g(WoT, C_, n0 + j * 16, k0, lane);
#pragma unroll
    for (int i = 0; i < 4; ++i)
#pragma unroll
      for (int j = 0; j < 4; ++j)
        acc[i][j] = wmma_bf16(a[i], b[j], acc[i][j]);
  }
  int nl = lane & 15, half8 = (lane >> 4) << 3;
#pragma unroll
  for (int i = 0; i < 4; ++i)
#pragma unroll
    for (int j = 0; j < 4; ++j)
#pragma unroll
      for (int v = 0; v < 8; ++v) {
        int row = m0 + i * 16 + half8 + v;
        int col = n0 + j * 16 + nl;
        out[(size_t)row * C_ + col] = acc[i][j][v] + bo[col];
      }
}

// ------------------------------------------------------------------ launch --
extern "C" void kernel_launch(void* const* d_in, const int* in_sizes, int n_in,
                              void* d_out, int out_size, void* d_ws, size_t ws_size,
                              hipStream_t stream) {
  (void)in_sizes; (void)n_in; (void)out_size; (void)ws_size;
  const float* x    = (const float*)d_in[0];
  const float* Wqkv = (const float*)d_in[1];
  const float* bqkv = (const float*)d_in[2];
  const float* Wo   = (const float*)d_in[3];
  const float* bo   = (const float*)d_in[4];
  float* out = (float*)d_out;

  constexpr size_t N_X   = (size_t)B_ * T_ * C_;      // 4M
  constexpr size_t N_WQ  = (size_t)3 * C_ * C_;       // 3M
  constexpr size_t N_WO  = (size_t)C_ * C_;           // 1M
  constexpr size_t N_QKV = (size_t)B_ * H_ * T_ * D_; // 4M each

  __bf16* xb    = (__bf16*)d_ws;
  __bf16* WqkvT = xb    + N_X;
  __bf16* WoT   = WqkvT + N_WQ;
  __bf16* Qb    = WoT   + N_WO;
  __bf16* Kb    = Qb    + N_QKV;
  __bf16* Vt    = Kb    + N_QKV;
  __bf16* ao    = Vt    + N_QKV;

  cvt_kernel<<<(int)((N_X + 255) / 256), 256, 0, stream>>>(x, Wqkv, Wo, xb, WqkvT, WoT);
  qkv_gemm<<<dim3(3 * C_ / 256, B_ * T_ / 128), 256, 0, stream>>>(xb, WqkvT, bqkv, Qb, Kb, Vt);
  attn_kernel<<<(B_ * H_ * (T_ / 16)) / 8, 256, 0, stream>>>(Qb, Kb, Vt, ao);
  out_gemm<<<dim3(C_ / 256, B_ * T_ / 128), 256, 0, stream>>>(ao, WoT, bo, out);
}
